// PAConv2_77876347011782
// MI455X (gfx1250) — compile-verified, matrix operand order checked by reading, not run on hardware
//
#include <hip/hip_runtime.h>
#include <hip/hip_bf16.h>

// Problem constants (match reference)
#define NFC 64
#define WW  256
#define HH  256
#define HWC 65536           // H*W
#define BB  4
#define PTOT (BB * HWC)     // 262144 pixels total
#define SLOPEC 0.2f

typedef __attribute__((ext_vector_type(16))) _Float16 v16h;
typedef __attribute__((ext_vector_type(8)))  float    v8f;

__device__ __forceinline__ float lrelu_f(float v) { return v >= 0.f ? v : SLOPEC * v; }
__device__ __forceinline__ float sigm_f(float v)  { return 1.f / (1.f + __expf(-v)); }
__device__ __forceinline__ int   clampi(int v, int lo, int hi) { return v < lo ? lo : (v > hi ? hi : v); }

// ---------------------------------------------------------------------------
// Shared WMMA core: D(64 out-ch x 16 px) = Wh(64x64) * Bfrag(64 x 16) + bias.
// A layout (16-bit, 16x32): lane<16 -> row M=lane, K in {0..7,16..23};
// lane>=16 -> same row, K in {8..15,24..31}.  B symmetric with N=lane&15.
// C/D: lane&15 = pixel column N; VGPR r = row M=r (+8 for lane>=16).
// HAS_BIAS is compile-time so the no-bias path has zero guards.
// ---------------------------------------------------------------------------
template <bool HAS_BIAS>
__device__ __forceinline__ void wmma_gemm_store(const _Float16* __restrict__ wlds,
                                                const float* __restrict__ bias,
                                                const v16h* __restrict__ bfr,
                                                float* __restrict__ outb,
                                                int q, int l16, int ln)
{
#pragma unroll
    for (int mt = 0; mt < 4; ++mt) {
        v8f acc;
#pragma unroll
        for (int r = 0; r < 8; ++r) {
            if constexpr (HAS_BIAS) acc[r] = bias[mt * 16 + l16 * 8 + r];
            else                    acc[r] = 0.0f;
        }
#pragma unroll
        for (int kk = 0; kk < 2; ++kk) {
            v16h afr;
            const _Float16* wrow = wlds + (mt * 16 + ln) * NFC + kk * 32 + l16 * 8;
#pragma unroll
            for (int e = 0; e < 8; ++e)  afr[e] = wrow[e];        // K = base+0..7
#pragma unroll
            for (int e = 8; e < 16; ++e) afr[e] = wrow[e + 8];    // K = base+16..23
            acc = __builtin_amdgcn_wmma_f32_16x16x32_f16(
                /*neg_a=*/false, afr, /*neg_b=*/false, bfr[kk],
                /*c_mod=*/(short)0, acc, /*reuse_a=*/false, /*reuse_b=*/false);
        }
#pragma unroll
        for (int r = 0; r < 8; ++r)
            outb[(size_t)(mt * 16 + l16 * 8 + r) * HWC + q] = acc[r];
    }
}

// ---------------------------------------------------------------------------
// Kernel A: condition branch.  Per pixel: flow(2), offset dy/dx, sigmoid mask.
// ---------------------------------------------------------------------------
__global__ __launch_bounds__(256) void cond_kernel(const float* __restrict__ cond,
                                                   const float* __restrict__ w1,
                                                   const float* __restrict__ b1,
                                                   const float* __restrict__ w2,
                                                   const float* __restrict__ b2,
                                                   const float* __restrict__ wcc,
                                                   const float* __restrict__ bcc,
                                                   const float* __restrict__ woff,
                                                   const float* __restrict__ boff,
                                                   float* __restrict__ params)
{
    __shared__ float w2l[NFC * NFC];
    for (int i = threadIdx.x; i < NFC * NFC; i += 256) w2l[i] = w2[i];
    __syncthreads();

    size_t idx = (size_t)blockIdx.x * 256 + threadIdx.x;       // < PTOT
    float cv = cond[idx];

    float fx = lrelu_f(wcc[0] * cv + bcc[0]);                  // flow x (channel 0)
    float fy = lrelu_f(wcc[1] * cv + bcc[1]);                  // flow y (channel 1)

    float f1[NFC];
#pragma unroll
    for (int j = 0; j < NFC; ++j) f1[j] = lrelu_f(w1[j] * cv + b1[j]);

    float om0 = boff[0], om1 = boff[1], om2 = boff[2];
    for (int o = 0; o < NFC; ++o) {
        float s = b2[o];
        const float* wr = &w2l[o * NFC];
#pragma unroll 8
        for (int j = 0; j < NFC; ++j) s = fmaf(wr[j], f1[j], s);
        s = lrelu_f(s);
        om0 = fmaf(woff[o],           s, om0);
        om1 = fmaf(woff[NFC + o],     s, om1);
        om2 = fmaf(woff[2 * NFC + o], s, om2);
    }
    params[idx]                      = fx;
    params[(size_t)PTOT + idx]       = fy;
    params[2 * (size_t)PTOT + idx]   = om0;                    // dy
    params[3 * (size_t)PTOT + idx]   = om1;                    // dx
    params[4 * (size_t)PTOT + idx]   = sigm_f(om2);            // mask
}

// ---------------------------------------------------------------------------
// Kernel B: deformable bilinear sample (zero-pad, *mask fused into corner
// weights) feeding WMMA with w_dcn.  One wave = 16 pixels.
// ---------------------------------------------------------------------------
__global__ __launch_bounds__(128) void deform_dcn_kernel(const float* __restrict__ x,
                                                         const float* __restrict__ wdcn,
                                                         const float* __restrict__ bdcn,
                                                         const float* __restrict__ params,
                                                         float* __restrict__ y)
{
    __shared__ _Float16 wlds[NFC * NFC];
    for (int i = threadIdx.x; i < NFC * NFC; i += 128) wlds[i] = (_Float16)wdcn[i];
    __syncthreads();

    int wave = threadIdx.x >> 5, lane = threadIdx.x & 31;
    int l16 = lane >> 4, ln = lane & 15;
    size_t tile = (size_t)blockIdx.x * 4 + wave;
    size_t p = tile * 16;
    int b = (int)(p >> 16);
    int q = (int)(p & (HWC - 1)) + ln;
    int h = q >> 8, w = q & (WW - 1);
    size_t gp = (size_t)b * HWC + q;

    float dy = params[2 * (size_t)PTOT + gp];
    float dx = params[3 * (size_t)PTOT + gp];
    float mk = params[4 * (size_t)PTOT + gp];

    float ys = (float)h + dy, xs = (float)w + dx;
    float y0f = floorf(ys), x0f = floorf(xs);
    float wy = ys - y0f, wx = xs - x0f;
    int y0 = (int)y0f, x0 = (int)x0f, y1 = y0 + 1, x1 = x0 + 1;

    float v00 = (((unsigned)y0 < 256u) && ((unsigned)x0 < 256u)) ? 1.f : 0.f;
    float v01 = (((unsigned)y0 < 256u) && ((unsigned)x1 < 256u)) ? 1.f : 0.f;
    float v10 = (((unsigned)y1 < 256u) && ((unsigned)x0 < 256u)) ? 1.f : 0.f;
    float v11 = (((unsigned)y1 < 256u) && ((unsigned)x1 < 256u)) ? 1.f : 0.f;
    int yc0 = clampi(y0, 0, 255), yc1 = clampi(y1, 0, 255);
    int xc0 = clampi(x0, 0, 255), xc1 = clampi(x1, 0, 255);
    int i00 = yc0 * WW + xc0, i01 = yc0 * WW + xc1;
    int i10 = yc1 * WW + xc0, i11 = yc1 * WW + xc1;
    float w00 = (1.f - wy) * (1.f - wx) * v00 * mk, w01 = (1.f - wy) * wx * v01 * mk;
    float w10 = wy * (1.f - wx) * v10 * mk,         w11 = wy * wx * v11 * mk;

    const float* xb = x + (size_t)b * NFC * HWC;
    __builtin_prefetch(xb + i00, 0, 0);          // global_prefetch_b8 hint
    v16h bfr[2];
#pragma unroll
    for (int kk = 0; kk < 2; ++kk) {
        const float* base = xb + (size_t)(kk * 32 + l16 * 8) * HWC;
#pragma unroll
        for (int e = 0; e < 16; ++e) {
            const float* xc = base + (size_t)((e < 8) ? e : (e + 8)) * HWC;
            float v = w00 * xc[i00] + w01 * xc[i01] + w10 * xc[i10] + w11 * xc[i11];
            bfr[kk][e] = (_Float16)v;
        }
    }
    wmma_gemm_store<true>(wlds, bdcn, bfr, y + (size_t)b * NFC * HWC, q, l16, ln);
}

// ---------------------------------------------------------------------------
// Kernel: generic 64x64 pointwise conv via WMMA (HAS_BIAS templated so the
// k3/k4 paths carry no per-load guards).
// ---------------------------------------------------------------------------
template <bool HAS_BIAS>
__global__ __launch_bounds__(128) void pw_wmma_kernel(const float* __restrict__ in,
                                                      const float* __restrict__ wpw,
                                                      const float* __restrict__ bias,
                                                      float* __restrict__ out)
{
    __shared__ _Float16 wlds[NFC * NFC];
    for (int i = threadIdx.x; i < NFC * NFC; i += 128) wlds[i] = (_Float16)wpw[i];
    __syncthreads();

    int wave = threadIdx.x >> 5, lane = threadIdx.x & 31;
    int l16 = lane >> 4, ln = lane & 15;
    size_t tile = (size_t)blockIdx.x * 4 + wave;
    size_t p = tile * 16;
    int b = (int)(p >> 16);
    int q = (int)(p & (HWC - 1)) + ln;

    const float* inb = in + (size_t)b * NFC * HWC;
    __builtin_prefetch(inb + q, 0, 0);
    v16h bfr[2];
#pragma unroll
    for (int kk = 0; kk < 2; ++kk) {
        const float* base = inb + (size_t)(kk * 32 + l16 * 8) * HWC + q;
#pragma unroll
        for (int e = 0; e < 8; ++e)  bfr[kk][e] = (_Float16)base[(size_t)e * HWC];
#pragma unroll
        for (int e = 8; e < 16; ++e) bfr[kk][e] = (_Float16)base[(size_t)(e + 8) * HWC];
    }
    wmma_gemm_store<HAS_BIAS>(wlds, bias, bfr, out + (size_t)b * NFC * HWC, q, l16, ln);
}

// ---------------------------------------------------------------------------
// Kernel C: y_flow = clamped bilinear of y along flow; gate = sigmoid(y+y_flow)
// ---------------------------------------------------------------------------
__global__ __launch_bounds__(256) void gate_kernel(const float* __restrict__ y,
                                                   const float* __restrict__ params,
                                                   float* __restrict__ gate)
{
    size_t idx = (size_t)blockIdx.x * 256 + threadIdx.x;   // < BB*NFC*HWC
    int q = (int)(idx & (HWC - 1));
    size_t bc = idx >> 16;                                  // b*NFC + c
    int b = (int)(bc >> 6);
    int h = q >> 8, w = q & (WW - 1);
    size_t gp = (size_t)b * HWC + q;

    float fx = params[gp];
    float fy = params[(size_t)PTOT + gp];
    float xs = fminf(fmaxf((float)w + fx, 0.f), 255.f);
    float ys = fminf(fmaxf((float)h + fy, 0.f), 255.f);
    float x0f = floorf(xs), y0f = floorf(ys);
    float wx = xs - x0f, wy = ys - y0f;
    int x0 = (int)x0f, y0 = (int)y0f;
    int x1 = x0 + 1 > 255 ? 255 : x0 + 1;
    int y1 = y0 + 1 > 255 ? 255 : y0 + 1;

    const float* yp = y + (bc << 16);
    float v00 = yp[y0 * WW + x0], v01 = yp[y0 * WW + x1];
    float v10 = yp[y1 * WW + x0], v11 = yp[y1 * WW + x1];
    float yfl = v00 * (1.f - wy) * (1.f - wx) + v01 * (1.f - wy) * wx
              + v10 * wy * (1.f - wx)         + v11 * wy * wx;
    gate[idx] = sigm_f(yp[q] + yfl);
}

// ---------------------------------------------------------------------------
// Kernel: depthwise 3x3 (SAME, zero pad); HAS_GATE templated multiply.
// ---------------------------------------------------------------------------
template <bool HAS_GATE>
__global__ __launch_bounds__(256) void dw3_kernel(const float* __restrict__ t,
                                                  const float* __restrict__ wdw,
                                                  const float* __restrict__ gate,
                                                  float* __restrict__ out)
{
    size_t idx = (size_t)blockIdx.x * 256 + threadIdx.x;   // < BB*NFC*HWC
    int q = (int)(idx & (HWC - 1));
    size_t bc = idx >> 16;
    int c = (int)(bc & 63);
    int h = q >> 8, w = q & (WW - 1);
    const float* tp = t + (bc << 16);
    const float* k = wdw + c * 9;
    float s = 0.f;
#pragma unroll
    for (int ky = 0; ky < 3; ++ky) {
        int yy = h + ky - 1;
        if ((unsigned)yy < 256u) {
#pragma unroll
            for (int kx = 0; kx < 3; ++kx) {
                int xx = w + kx - 1;
                if ((unsigned)xx < 256u) s = fmaf(k[ky * 3 + kx], tp[yy * WW + xx], s);
            }
        }
    }
    if constexpr (HAS_GATE) s *= gate[idx];
    out[idx] = s;
}

// ---------------------------------------------------------------------------
extern "C" void kernel_launch(void* const* d_in, const int* in_sizes, int n_in,
                              void* d_out, int out_size, void* d_ws, size_t ws_size,
                              hipStream_t stream)
{
    (void)in_sizes; (void)n_in; (void)out_size; (void)ws_size;
    const float* x      = (const float*)d_in[0];
    const float* cond   = (const float*)d_in[1];
    const float* w_oc1  = (const float*)d_in[2];
    const float* b_oc1  = (const float*)d_in[3];
    const float* w_oc2  = (const float*)d_in[4];
    const float* b_oc2  = (const float*)d_in[5];
    const float* w_cc   = (const float*)d_in[6];
    const float* b_cc   = (const float*)d_in[7];
    const float* w_off  = (const float*)d_in[8];
    const float* b_off  = (const float*)d_in[9];
    const float* w_dcn  = (const float*)d_in[10];
    const float* b_dcn  = (const float*)d_in[11];
    const float* w_k3pw = (const float*)d_in[12];
    const float* w_k3dw = (const float*)d_in[13];
    const float* w_k4pw = (const float*)d_in[14];
    const float* w_k4dw = (const float*)d_in[15];

    float* params = (float*)d_ws;                       // 5 * PTOT floats
    float* buf1   = params + 5 * (size_t)PTOT;          // NFC * PTOT floats
    float* buf2   = buf1   + (size_t)NFC * PTOT;        // NFC * PTOT floats

    const int nWGpw = PTOT / 64;                        // 4096 (4 waves * 16 px)
    const int nWGpx = (NFC * PTOT) / 256;               // 65536

    // A: condition branch -> params
    cond_kernel<<<PTOT / 256, 256, 0, stream>>>(cond, w_oc1, b_oc1, w_oc2, b_oc2,
                                                w_cc, b_cc, w_off, b_off, params);
    // B: deformable sample + w_dcn (WMMA) -> buf1 = y
    deform_dcn_kernel<<<nWGpw, 128, 0, stream>>>(x, w_dcn, b_dcn, params, buf1);
    // C: gate = sigmoid(y + bilinear(y, flow)) -> buf2
    gate_kernel<<<nWGpx, 256, 0, stream>>>(buf1, params, buf2);
    // D1: t3 = x * w_k3_pw (WMMA, no bias) -> buf1 (y dead)
    pw_wmma_kernel<false><<<nWGpw, 128, 0, stream>>>(x, w_k3pw, nullptr, buf1);
    // D2: out1 = dw3(t3) * gate, in place over gate -> buf2
    dw3_kernel<true><<<nWGpx, 256, 0, stream>>>(buf1, w_k3dw, buf2, buf2);
    // E1: t4 = out1 * w_k4_pw (WMMA, no bias) -> buf1
    pw_wmma_kernel<false><<<nWGpw, 128, 0, stream>>>(buf2, w_k4pw, nullptr, buf1);
    // E2: out = dw3(t4) -> d_out
    dw3_kernel<false><<<nWGpx, 256, 0, stream>>>(buf1, w_k4dw, nullptr, (float*)d_out);
}
